// InfiniKANTransformer_16192026706525
// MI455X (gfx1250) — matrix operand order repaired and checked by reading, not compiled, not run on hardware
//
#include <hip/hip_runtime.h>
#include <hip/hip_bf16.h>
#include <math.h>

// ---- problem constants -------------------------------------------------
static constexpr int   Vv = 32000;
static constexpr int   Dd = 1024;
static constexpr int   Ll = 6;
static constexpr int   Hh = 16;
static constexpr int   DHh = 64;
static constexpr int   Bb = 2;
static constexpr int   Nn = 1024;
static constexpr int   FF = 4096;
static constexpr int   Mrows = Bb * Nn;          // 2048

// ---- CDNA5 WMMA types --------------------------------------------------
typedef __attribute__((ext_vector_type(16))) __bf16 v16bf;
typedef __attribute__((ext_vector_type(8)))  float  v8f;

__device__ __forceinline__ unsigned short bfc(float f) {   // hw f32->bf16 (RNE)
    __bf16 h = (__bf16)f;
    return __builtin_bit_cast(unsigned short, h);
}
__device__ __forceinline__ v16bf ld_frag(const unsigned short* p) {
    return *(const v16bf*)p;       // 32B contiguous -> 2x ds_load_b128
}
__device__ __forceinline__ unsigned lds_off(const void* p) {
    return (unsigned)(uintptr_t)p; // LDS aperture: low 32 bits = LDS address
}
__device__ __forceinline__ float elup1(float x) {          // elu(x)+1
    return x > 0.0f ? x + 1.0f : __expf(x);
}
__device__ __forceinline__ float gelu_exact(float x) {
    return 0.5f * x * (1.0f + erff(x * 0.70710678118654752f));
}

// =======================================================================
// 1) embedding gather
// =======================================================================
__global__ __launch_bounds__(256) void embed_kernel(const int* __restrict__ tokens,
                                                    const float* __restrict__ emb,
                                                    float* __restrict__ x) {
    int row = blockIdx.x;
    int tok = tokens[row];
    const float* src = emb + (size_t)tok * Dd;
    float* dst = x + (size_t)row * Dd;
    for (int i = threadIdx.x; i < Dd; i += 256) dst[i] = src[i];
}

// =======================================================================
// 2) RMS-norm (fp32 in -> bf16 out), row-per-block
// =======================================================================
__global__ __launch_bounds__(256) void rmsnorm_kernel(const float* __restrict__ x,
                                                      const float* __restrict__ gamma,
                                                      unsigned short* __restrict__ out) {
    __shared__ float red[8];
    int row = blockIdx.x;
    const float* xr = x + (size_t)row * Dd;
    float ss = 0.0f;
    for (int i = threadIdx.x; i < Dd; i += 256) { float v = xr[i]; ss += v * v; }
#pragma unroll
    for (int m = 1; m < 32; m <<= 1) ss += __shfl_xor(ss, m, 32);
    int wave = threadIdx.x >> 5, lane = threadIdx.x & 31;
    if (lane == 0) red[wave] = ss;
    __syncthreads();
    if (wave == 0) {
        float v = (lane < 8) ? red[lane] : 0.0f;
#pragma unroll
        for (int m = 1; m < 8; m <<= 1) v += __shfl_xor(v, m, 32);
        if (lane == 0) red[0] = v;
    }
    __syncthreads();
    float inv = rsqrtf(red[0] + 1e-12f) * 32.0f;        // sqrt(D)=32
    unsigned short* orow = out + (size_t)row * Dd;
    for (int i = threadIdx.x; i < Dd; i += 256)
        orow[i] = bfc(xr[i] * inv * gamma[i]);
}

// =======================================================================
// 3) bf16 WMMA GEMM  C[M x N] = A[M x K](bf16) @ B[K x N](f32->bf16)
//    Block tile 128x128, 8 waves (2x4), wave tile 64x32, K-step 32.
//    A tile staged via global_load_async_to_lds_b128 into row-major LDS
//    (k contiguous, half h reads k = 16h..16h+15) so A/B fragments are single 32B LDS reads.
//    B tile staged transposed (fragment-major) with hw bf16 converts.
//    EPI: 0 = f32 store, 1 = f32 + residual R, 2 = gelu -> bf16
// =======================================================================
template <int EPI>
__global__ __launch_bounds__(256) void gemm_bf16_kernel(const unsigned short* __restrict__ A,
                                                        const float* __restrict__ B,
                                                        float* __restrict__ Cf,
                                                        unsigned short* __restrict__ Cb,
                                                        const float* __restrict__ R,
                                                        int Ntot, int K) {
    __shared__ __align__(64) unsigned short sA[128][32];   // [row][k], k contiguous
    __shared__ __align__(64) unsigned short sBt[128][32];  // [col][k], fragment-major

    const int tid  = threadIdx.x;
    const int wave = tid >> 5, lane = tid & 31;
    const int rlane = lane & 15;
    const int h8   = (lane >> 4) * 8;        // D-layout row offset
    const int hl16 = (lane >> 4) * 16;       // A frag k-offset
    const int kb16 = (lane >> 4) * 16;       // B frag k-offset
    const int bm = blockIdx.y * 128, bn = blockIdx.x * 128;
    const int wm = (wave >> 2) * 64, wn = (wave & 3) * 32;

    v8f acc[4][2];
#pragma unroll
    for (int i = 0; i < 4; ++i)
#pragma unroll
        for (int j = 0; j < 2; ++j) acc[i][j] = {};

    // A tile async-load map: 16 bf16 per thread, two b128 transfers.
    const int arow = tid >> 1, acs = (tid & 1) * 16;
    const int ad0  = acs;                                 // contiguous dests
    const unsigned ldsA0 = lds_off(&sA[arow][ad0]);
    const unsigned ldsA1 = lds_off(&sA[arow][ad0 + 8]);
    // B tile map: 2 adjacent k-rows x 8 cols per thread, packed u32 stores.
    const int br2 = (tid >> 4) * 2, bc8 = (tid & 15) * 8;

    for (int k0 = 0; k0 < K; k0 += 32) {
        __syncthreads();
        // ---- stage A (bf16) via CDNA5 async global->LDS DMA -----------
        {
            const unsigned short* g0 = A + (size_t)(bm + arow) * K + k0 + acs;
            asm volatile("global_load_async_to_lds_b128 %0, %1, off"
                         :: "v"(ldsA0), "v"(g0) : "memory");
            asm volatile("global_load_async_to_lds_b128 %0, %1, off"
                         :: "v"(ldsA1), "v"(g0 + 8) : "memory");
        }
        // ---- stage B (f32 -> bf16, transposed) -------------------------
        {
            const float* s0p = B + (size_t)(k0 + br2) * Ntot + bn + bc8;
            const float* s1p = s0p + Ntot;
            if (k0 + 32 < K) __builtin_prefetch(s0p + (size_t)32 * Ntot, 0, 1);
#pragma unroll
            for (int i = 0; i < 8; ++i) {
                unsigned pk = (unsigned)bfc(s0p[i]) | ((unsigned)bfc(s1p[i]) << 16);
                *(unsigned*)&sBt[bc8 + i][br2] = pk;
            }
        }
        asm volatile("s_wait_asynccnt 0x0" ::: "memory");
        __syncthreads();

        FragBFDECL:;
        v16bf af[4], bf[2];
#pragma unroll
        for (int i2 = 0; i2 < 4; ++i2)
            af[i2] = ld_frag(&sA[wm + i2 * 16 + rlane][hl16]);
#pragma unroll
        for (int j2 = 0; j2 < 2; ++j2)
            bf[j2] = ld_frag(&sBt[wn + j2 * 16 + rlane][kb16]);
#pragma unroll
        for (int i2 = 0; i2 < 4; ++i2)
#pragma unroll
            for (int j2 = 0; j2 < 2; ++j2)
                acc[i2][j2] = __builtin_amdgcn_wmma_f32_16x16x32_bf16(
                    false, af[i2], false, bf[j2], (short)0, acc[i2][j2],
                    false, false);
    }

    // epilogue (D layout: row = h8+v, col = rlane within 16x16 tile)
#pragma unroll
    for (int i2 = 0; i2 < 4; ++i2)
#pragma unroll
        for (int j2 = 0; j2 < 2; ++j2)
#pragma unroll
            for (int v = 0; v < 8; ++v) {
                int r = bm + wm + i2 * 16 + h8 + v;
                int c = bn + wn + j2 * 16 + rlane;
                size_t idx = (size_t)r * Ntot + c;
                float val = acc[i2][j2][v];
                if constexpr (EPI == 0) Cf[idx] = val;
                if constexpr (EPI == 1) Cf[idx] = val + R[idx];
                if constexpr (EPI == 2) Cb[idx] = bfc(gelu_exact(val));
            }
}

// =======================================================================
// 4) QKV split + rotary + head transpose + elu feature maps
// =======================================================================
__global__ __launch_bounds__(256) void rope_heads_kernel(const float* __restrict__ qkv,
                                                         float* __restrict__ qh,
                                                         float* __restrict__ kh,
                                                         float* __restrict__ vh,
                                                         float* __restrict__ qf,
                                                         float* __restrict__ kf) {
    int idx = blockIdx.x * 256 + threadIdx.x;        // B*N*H*32 items
    int p = idx & 31;
    int h = (idx >> 5) & (Hh - 1);
    int n = (idx >> 9) & (Nn - 1);
    int b = idx >> 19;
    const float* base = qkv + (size_t)(b * Nn + n) * (3 * Dd);
    int doff = h * DHh + 2 * p;
    float q0 = base[doff],           q1 = base[doff + 1];
    float k0 = base[Dd + doff],      k1 = base[Dd + doff + 1];
    float v0 = base[2 * Dd + doff],  v1 = base[2 * Dd + doff + 1];
    float freq = (float)n * __powf(10000.0f, -(float)p / 32.0f);
    float s, c;
    __sincosf(freq, &s, &c);
    float qr0 = q0 * c - q1 * s, qr1 = q1 * c + q0 * s;
    float kr0 = k0 * c - k1 * s, kr1 = k1 * c + k0 * s;
    size_t ho = (((size_t)(b * Hh + h)) * Nn + n) * DHh + 2 * p;
    qh[ho] = qr0; qh[ho + 1] = qr1;
    kh[ho] = kr0; kh[ho + 1] = kr1;
    vh[ho] = v0;  vh[ho + 1] = v1;
    qf[ho] = elup1(qr0); qf[ho + 1] = elup1(qr1);
    kf[ho] = elup1(kr0); kf[ho + 1] = elup1(kr1);
}

// =======================================================================
// 5) flash attention, one (b,h, 128-row q-chunk) per block, WMMA bf16.
//    V staged transposed so P@V fragments are contiguous 32B LDS reads.
// =======================================================================
__global__ __launch_bounds__(256) void attn_kernel(const float* __restrict__ qh,
                                                   const float* __restrict__ kh,
                                                   const float* __restrict__ vh,
                                                   float* __restrict__ outO) {
    const int QC = Nn / 128;
    int qc = blockIdx.x % QC;
    int bh = blockIdx.x / QC;
    const float* Q = qh + ((size_t)bh * Nn + qc * 128) * DHh;
    const float* K = kh + (size_t)bh * Nn * DHh;
    const float* V = vh + (size_t)bh * Nn * DHh;
    float* O = outO + ((size_t)bh * Nn + qc * 128) * DHh;

    __shared__ __align__(64) unsigned short sQ[128][64];
    __shared__ __align__(64) unsigned short sK[32][64];     // [j][k], rows contiguous
    __shared__ __align__(64) unsigned short sVt[64][32];    // [d][j], fragment-major
    __shared__ __align__(64) unsigned short sP[8][16][32];

    const int tid = threadIdx.x;
    const int wave = tid >> 5, lane = tid & 31;
    const int rlane = lane & 15;
    const int h8 = (lane >> 4) * 8;
    const int kb16 = (lane >> 4) * 16;

    for (int i = tid; i < 128 * 64; i += 256)
        sQ[i >> 6][i & 63] = bfc(Q[i] * 0.125f);             // DH^-0.5 folded
    __syncthreads();

    // Q A-fragments (k = 0..31, 32..63) for this wave's 16 rows
    union { v16bf v; unsigned short u[16]; } qa[2];
#pragma unroll
    for (int m = 0; m < 2; ++m)
#pragma unroll
        for (int i = 0; i < 8; ++i) {
            qa[m].u[i]     = sQ[wave * 16 + rlane][m * 32 + kb16 + i];
            qa[m].u[8 + i] = sQ[wave * 16 + rlane][m * 32 + kb16 + 8 + i];
        }

    float mrow[8], lrow[8];
    v8f oacc[4];
#pragma unroll
    for (int v = 0; v < 8; ++v) { mrow[v] = -3.0e38f; lrow[v] = 0.0f; }
#pragma unroll
    for (int t = 0; t < 4; ++t) oacc[t] = {};

    const int vd = tid & 63, vjb = (tid >> 6) * 8;           // V staging map

    for (int j0 = 0; j0 < Nn; j0 += 32) {
        __syncthreads();
        for (int i = tid; i < 32 * 64; i += 256)
            sK[i >> 6][i & 63] = bfc(K[(size_t)j0 * 64 + i]);
#pragma unroll
        for (int jj = 0; jj < 8; jj += 2) {
            float a0 = V[(size_t)(j0 + vjb + jj) * 64 + vd];
            float a1 = V[(size_t)(j0 + vjb + jj + 1) * 64 + vd];
            *(unsigned*)&sVt[vd][vjb + jj] =
                (unsigned)bfc(a0) | ((unsigned)bfc(a1) << 16);
        }
        __syncthreads();

        // S = Q @ K^T : two 16x16 col-tiles, K-dim 64 = 2 wmma each
        v8f s0 = {}, s1 = {};
#pragma unroll
        for (int m = 0; m < 2; ++m) {
            v16bf kb0 = ld_frag(&sK[rlane][m * 32 + kb16]);
            v16bf kb1 = ld_frag(&sK[16 + rlane][m * 32 + kb16]);
            s0 = __builtin_amdgcn_wmma_f32_16x16x32_bf16(false, qa[m].v, false, kb0,
                                                         (short)0, s0, false, false);
            s1 = __builtin_amdgcn_wmma_f32_16x16x32_bf16(false, qa[m].v, false, kb1,
                                                         (short)0, s1, false, false);
        }

        // online softmax: row stats over the 16-lane half holding each row
#pragma unroll
        for (int v = 0; v < 8; ++v) {
            float mx = fmaxf(s0[v], s1[v]);
#pragma unroll
            for (int msk = 1; msk < 16; msk <<= 1) mx = fmaxf(mx, __shfl_xor(mx, msk, 32));
            float mnew = fmaxf(mrow[v], mx);
            float alpha = __expf(mrow[v] - mnew);
            float p0 = __expf(s0[v] - mnew);
            float p1 = __expf(s1[v] - mnew);
            float rs = p0 + p1;
#pragma unroll
            for (int msk = 1; msk < 16; msk <<= 1) rs += __shfl_xor(rs, msk, 32);
            lrow[v] = lrow[v] * alpha + rs;
            mrow[v] = mnew;
#pragma unroll
            for (int t = 0; t < 4; ++t) oacc[t][v] *= alpha;
            sP[wave][h8 + v][rlane]      = bfc(p0);
            sP[wave][h8 + v][16 + rlane] = bfc(p1);
        }
        // wave-private LDS transpose of P: wait DS, then continue
        asm volatile("s_wait_dscnt 0x0" ::: "memory");
        __builtin_amdgcn_wave_barrier();

        // O += P(16x32) @ V(32x64)
        union { v16bf v; unsigned short u[16]; } pa;
#pragma unroll
        for (int i = 0; i < 8; ++i) {
            pa.u[i]     = sP[wave][rlane][kb16 + i];
            pa.u[8 + i] = sP[wave][rlane][kb16 + 8 + i];
        }
#pragma unroll
        for (int t = 0; t < 4; ++t) {
            v16bf vb = ld_frag(&sVt[t * 16 + rlane][kb16]);
            oacc[t] = __builtin_amdgcn_wmma_f32_16x16x32_bf16(false, pa.v, false, vb,
                                                              (short)0, oacc[t], false, false);
        }
    }

    // write normalized output
#pragma unroll
    for (int t = 0; t < 4; ++t)
#pragma unroll
        for (int v = 0; v < 8; ++v) {
            int r = wave * 16 + h8 + v;
            int c = t * 16 + rlane;
            O[(size_t)r * DHh + c] = oacc[t][v] / lrow[v];
        }
}

// =======================================================================
// 6) memory retrieval + gate mix -> combined heads (bf16 [B,N,D])
// =======================================================================
__global__ __launch_bounds__(256) void retgate_kernel(const float* __restrict__ qf,
                                                      const float* __restrict__ attnO,
                                                      const float* __restrict__ memkv,
                                                      const float* __restrict__ memnorm,
                                                      const float* __restrict__ gates,
                                                      unsigned short* __restrict__ comb) {
    __shared__ float skv[64][64];
    __shared__ float snorm[64];
    int nc = blockIdx.x & (Nn / 4 - 1);
    int bh = blockIdx.x >> 8;
    int h = bh & (Hh - 1), b = bh >> 4;
    int tid = threadIdx.x;
    for (int i = tid; i < 64 * 64; i += 256) skv[i >> 6][i & 63] = memkv[(size_t)bh * 4096 + i];
    if (tid < 64) snorm[tid] = memnorm[(size_t)bh * 64 + tid];
    __syncthreads();

    int e = tid & 63, nl = tid >> 6;
    int n = nc * 4 + nl;
    const float* q = qf + ((size_t)bh * Nn + n) * DHh;
    float num = 0.0f, den = 0.0f;
#pragma unroll 8
    for (int d = 0; d < 64; ++d) {
        float qv = q[d];
        num += qv * skv[d][e];
        den += qv * snorm[d];
    }
    float mem = num / fmaxf(den, 1e-10f);
    float g = 1.0f / (1.0f + __expf(-gates[h]));
    float a = attnO[((size_t)bh * Nn + n) * DHh + e];
    float val = a * g + mem * (1.0f - g);
    comb[(size_t)(b * Nn + n) * Dd + h * DHh + e] = bfc(val);
}

// =======================================================================
// 7) memory update: delta_v retrieve, v2 = v - delta, kv += kf^T @ v2,
//    norm += sum_n kf.   One block per (b,h), fp32 for precision.
// =======================================================================
__global__ __launch_bounds__(256) void kvupdate_kernel(const float* __restrict__ kf,
                                                       const float* __restrict__ vh,
                                                       const float* __restrict__ memkv,
                                                       const float* __restrict__ memnorm,
                                                       float* __restrict__ outkv,
                                                       float* __restrict__ outnorm) {
    __shared__ float skv[64][64];
    __shared__ float snorm[64];
    __shared__ float kfc[16][64];
    __shared__ float v2c[16][64];
    int bh = blockIdx.x;
    int tid = threadIdx.x;
    for (int i = tid; i < 64 * 64; i += 256) skv[i >> 6][i & 63] = memkv[(size_t)bh * 4096 + i];
    if (tid < 64) snorm[tid] = memnorm[(size_t)bh * 64 + tid];
    __syncthreads();

    int d = tid >> 2, e0 = (tid & 3) * 16;
    float acc[16];
#pragma unroll
    for (int j = 0; j < 16; ++j) acc[j] = 0.0f;
    float nacc = 0.0f;

    for (int n0 = 0; n0 < Nn; n0 += 16) {
        __syncthreads();
#pragma unroll
        for (int rep = 0; rep < 4; ++rep) {
            int w = tid + rep * 256;
            int nl = w >> 6, e = w & 63;
            kfc[nl][e] = kf[((size_t)bh * Nn + n0 + nl) * DHh + e];
        }
        __syncthreads();
#pragma unroll
        for (int rep = 0; rep < 4; ++rep) {
            int w = tid + rep * 256;
            int nl = w >> 6, e = w & 63;
            float num = 0.0f, den = 0.0f;
#pragma unroll 8
            for (int d2 = 0; d2 < 64; ++d2) {
                float kq = kfc[nl][d2];
                num += kq * skv[d2][e];
                den += kq * snorm[d2];
            }
            v2c[nl][e] = vh[((size_t)bh * Nn + n0 + nl) * DHh + e] - num / fmaxf(den, 1e-10f);
        }
        __syncthreads();
#pragma unroll
        for (int nl = 0; nl < 16; ++nl) {
            float kd = kfc[nl][d];
            if ((tid & 3) == 0) nacc += kd;
#pragma unroll
            for (int j = 0; j < 16; ++j) acc[j] += kd * v2c[nl][e0 + j];
        }
    }
    __syncthreads();
#pragma unroll
    for (int j = 0; j < 16; ++j)
        outkv[(size_t)bh * 4096 + d * 64 + e0 + j] = acc[j] + skv[d][e0 + j];
    if ((tid & 3) == 0) outnorm[(size_t)bh * 64 + d] = nacc + snorm[d];
}

// =======================================================================
// host-side launcher
// =======================================================================
extern "C" void kernel_launch(void* const* d_in, const int* in_sizes, int n_in,
                              void* d_out, int out_size, void* d_ws, size_t ws_size,
                              hipStream_t stream) {
    (void)in_sizes; (void)n_in; (void)out_size; (void)ws_size;
    const int*   tokens      = (const int*)d_in[0];
    const float* past_kv     = (const float*)d_in[1];
    const float* past_norm   = (const float*)d_in[2];
    const float* emb         = (const float*)d_in[3];
    const float* attn_gamma  = (const float*)d_in[4];
    const float* Wqkv        = (const float*)d_in[5];
    const float* Wout        = (const float*)d_in[6];
    const float* gates       = (const float*)d_in[7];
    const float* ff_gamma    = (const float*)d_in[8];
    const float* W1          = (const float*)d_in[9];
    const float* W2          = (const float*)d_in[10];
    const float* final_gamma = (const float*)d_in[11];
    const float* Wlogits     = (const float*)d_in[12];

    float* out_logits = (float*)d_out;
    float* out_kv     = out_logits + (size_t)Mrows * Vv;
    float* out_norm   = out_kv + (size_t)Ll * Bb * Hh * DHh * DHh;

    auto au = [](size_t x) { return (x + 255) & ~(size_t)255; };
    char* w = (char*)d_ws;
    float* x = (float*)w;                 w += au((size_t)Mrows * Dd * 4);
    unsigned short* bfa = (unsigned short*)w; w += au((size_t)Mrows * Dd * 2);
    float* qkvb = (float*)w;              w += au((size_t)Mrows * 3 * Dd * 4);
    float* qh = (float*)w;                w += au((size_t)Bb * Hh * Nn * DHh * 4);
    float* kh = (float*)w;                w += au((size_t)Bb * Hh * Nn * DHh * 4);
    float* vh = (float*)w;                w += au((size_t)Bb * Hh * Nn * DHh * 4);
    float* qf = (float*)w;                w += au((size_t)Bb * Hh * Nn * DHh * 4);
    float* kf = (float*)w;                w += au((size_t)Bb * Hh * Nn * DHh * 4);
    float* attnO = (float*)w;             w += au((size_t)Bb * Hh * Nn * DHh * 4);
    unsigned short* comb = (unsigned short*)w; w += au((size_t)Mrows * Dd * 2);
    unsigned short* ff1 = (unsigned short*)w;  w += au((size_t)Mrows * FF * 2);

    dim3 blk(256);

    embed_kernel<<<Mrows, blk, 0, stream>>>(tokens, emb, x);

    for (int l = 0; l < Ll; ++l) {
        const float* pkv = past_kv + (size_t)l * Bb * Hh * DHh * DHh;
        const float* pnm = past_norm + (size_t)l * Bb * Hh * DHh;

        rmsnorm_kernel<<<Mrows, blk, 0, stream>>>(x, attn_gamma + (size_t)l * Dd, bfa);
        gemm_bf16_kernel<0><<<dim3(3 * Dd / 128, Mrows / 128), blk, 0, stream>>>(
            bfa, Wqkv + (size_t)l * Dd * 3 * Dd, qkvb, nullptr, nullptr, 3 * Dd, Dd);
        rope_heads_kernel<<<(Bb * Nn * Hh * 32) / 256, blk, 0, stream>>>(qkvb, qh, kh, vh, qf, kf);
        attn_kernel<<<Bb * Hh * (Nn / 128), blk, 0, stream>>>(qh, kh, vh, attnO);
        retgate_kernel<<<Bb * Hh * (Nn / 4), blk, 0, stream>>>(qf, attnO, pkv, pnm,
                                                               gates + (size_t)l * Hh, comb);
        gemm_bf16_kernel<1><<<dim3(Dd / 128, Mrows / 128), blk, 0, stream>>>(
            comb, Wout + (size_t)l * Dd * Dd, x, nullptr, x, Dd, Dd);
        rmsnorm_kernel<<<Mrows, blk, 0, stream>>>(x, ff_gamma + (size_t)l * Dd, bfa);
        gemm_bf16_kernel<2><<<dim3(FF / 128, Mrows / 128), blk, 0, stream>>>(
            bfa, W1 + (size_t)l * Dd * FF, nullptr, ff1, nullptr, FF, Dd);
        gemm_bf16_kernel<1><<<dim3(Dd / 128, Mrows / 128), blk, 0, stream>>>(
            ff1, W2 + (size_t)l * FF * Dd, x, nullptr, x, Dd, FF);
        kvupdate_kernel<<<Bb * Hh, blk, 0, stream>>>(
            kf, vh, pkv, pnm,
            out_kv + (size_t)l * Bb * Hh * DHh * DHh,
            out_norm + (size_t)l * Bb * Hh * DHh);
    }

    rmsnorm_kernel<<<Mrows, blk, 0, stream>>>(x, final_gamma, bfa);
    gemm_bf16_kernel<0><<<dim3(Vv / 128, Mrows / 128), blk, 0, stream>>>(
        bfa, Wlogits, out_logits, nullptr, nullptr, Vv, Dd);
}
